// TreeLstmDecoderComplete_7524782702720
// MI455X (gfx1250) — matrix operand order repaired and checked by reading, not compiled, not run on hardware
//
#include <hip/hip_runtime.h>
#include <hip/hip_bf16.h>

// Problem dims (fixed by reference)
#define TSTEPS 32
#define MROWS  512
#define DDIM   256
#define ELDIM  128
#define VRDIM  600
#define VNDIM  8000
#define NNODES (TSTEPS * MROWS)

typedef __bf16 bf16_t;
typedef __attribute__((ext_vector_type(16))) __bf16 v16bf;
typedef __attribute__((ext_vector_type(8)))  float  v8f;

union BF16Frag { v16bf v; unsigned short u[16]; };

__device__ __forceinline__ unsigned short f32_to_bf16_bits(float x) {
  unsigned int u = __builtin_bit_cast(unsigned int, x);
  u = (u + 0x7FFFu + ((u >> 16) & 1u)) >> 16;   // round-to-nearest-even
  return (unsigned short)u;
}

// ---- WMMA fragment loaders (16-bit, contiguous k per lane half, same for A and B, wave32) --------------------
// A 16x32 bf16 (row-major bf16, ld = K):
// lane L: row = m0 + (L&15); elements = A[row, k0 + (L>>4)*16 + 0..15]
__device__ __forceinline__ v16bf load_a_frag(const bf16_t* __restrict__ A, int lda,
                                             int m0, int k0, int lane) {
  const int row = m0 + (lane & 15);
  const int kh  = k0 + ((lane >> 4) << 4);
  const unsigned short* p = (const unsigned short*)(A + (size_t)row * lda + kh);
  BF16Frag f;
#pragma unroll
  for (int i = 0; i < 8; ++i) {
    f.u[i]     = p[i];
    f.u[8 + i] = p[8 + i];
  }
  return f.v;
}

// B 32x16 bf16 (weights stored N x K row-major bf16, ld = K):
// lane L: col = n0 + (L&15); elements = B[col, k0 + (L>>4)*16 + 0..15]
__device__ __forceinline__ v16bf load_b_frag(const bf16_t* __restrict__ B, int ldb,
                                             int n0, int k0, int lane) {
  const int col = n0 + (lane & 15);
  const int kk  = k0 + ((lane >> 4) << 4);
  const unsigned short* p = (const unsigned short*)(B + (size_t)col * ldb + kk);
  BF16Frag f;
#pragma unroll
  for (int i = 0; i < 16; ++i) f.u[i] = p[i];
  return f.v;
}

__device__ __forceinline__ v16bf load_b_frag_clamped(const bf16_t* __restrict__ B,
                                                     int ldb, int n0, int k0,
                                                     int lane, int Ndim) {
  int col = n0 + (lane & 15);
  if (col >= Ndim) col = Ndim - 1;          // clamp, results masked at store
  const int kk = k0 + ((lane >> 4) << 4);
  const unsigned short* p = (const unsigned short*)(B + (size_t)col * ldb + kk);
  BF16Frag f;
#pragma unroll
  for (int i = 0; i < 16; ++i) f.u[i] = p[i];
  return f.v;
}

__device__ __forceinline__ void wmma4(const v16bf& a, const v16bf b[4], v8f acc[4]) {
#pragma unroll
  for (int j = 0; j < 4; ++j)
    acc[j] = __builtin_amdgcn_wmma_f32_16x16x32_bf16(
        false, a, false, b[j], (short)0, acc[j], false, false);
}

// Full-width (N-tile entirely in range) pipelined inner product.
// Ping-pong double buffering, unrolled by 2 k-steps (requires K % 64 == 0,
// true for all K here: 256 and 128). Loads write straight into the inactive
// buffer -> no rotation movs, and next-phase VMEM overlaps current WMMAs.
__device__ __forceinline__ void gemm_panel_fast(const bf16_t* __restrict__ A,
                                                const bf16_t* __restrict__ B, int K,
                                                int m0, int nbase, int lane,
                                                v8f acc[4]) {
  __builtin_prefetch(A + (size_t)(m0 + (lane & 15)) * K, 0, 1);
  v16bf a0 = load_a_frag(A, K, m0, 0, lane);
  v16bf b0[4];
#pragma unroll
  for (int j = 0; j < 4; ++j) b0[j] = load_b_frag(B, K, nbase + j * 16, 0, lane);

  for (int k0 = 0; k0 < K; k0 += 64) {
    // phase 0: prefetch k0+32 into buffer 1, compute on buffer 0
    v16bf a1 = load_a_frag(A, K, m0, k0 + 32, lane);
    v16bf b1[4];
#pragma unroll
    for (int j = 0; j < 4; ++j) b1[j] = load_b_frag(B, K, nbase + j * 16, k0 + 32, lane);
    wmma4(a0, b0, acc);
    // phase 1: prefetch k0+64 into buffer 0 (if any), compute on buffer 1
    const bool more = (k0 + 64 < K);         // scalar condition
    if (more) {
      a0 = load_a_frag(A, K, m0, k0 + 64, lane);
#pragma unroll
      for (int j = 0; j < 4; ++j)
        b0[j] = load_b_frag(B, K, nbase + j * 16, k0 + 64, lane);
    }
    wmma4(a1, b1, acc);
  }
}

// Edge path: per-16-column scalar guards + clamped B loads.
__device__ __forceinline__ void gemm_panel_edge(const bf16_t* __restrict__ A,
                                                const bf16_t* __restrict__ B, int K,
                                                int m0, int nbase, int lane, int Ndim,
                                                v8f acc[4]) {
  for (int k0 = 0; k0 < K; k0 += 32) {
    v16bf a = load_a_frag(A, K, m0, k0, lane);
#pragma unroll
    for (int j = 0; j < 4; ++j) {
      const int n0 = nbase + j * 16;         // sgpr + imm: stays scalar
      if (n0 < Ndim) {
        v16bf b = load_b_frag_clamped(B, K, n0, k0, lane, Ndim);
        acc[j] = __builtin_amdgcn_wmma_f32_16x16x32_bf16(
            false, a, false, b, (short)0, acc[j], false, false);
      }
    }
  }
}

// ---- Generic fused GEMM: C = act(A1@B1^T [+ A2@B2^T] [+ bias(col)] [+ rowoff(row)])
// A: bf16 M x K row-major; B: bf16 N x K row-major. One wave -> 16x64 C tile.
__global__ void __launch_bounds__(128)
k_gemm(const bf16_t* __restrict__ A1, const bf16_t* __restrict__ B1, int K1,
       const bf16_t* __restrict__ A2, const bf16_t* __restrict__ B2, int K2,
       const float* __restrict__ bias, const float* __restrict__ rowoff,
       float* __restrict__ C, bf16_t* __restrict__ Cb,
       int Mdim, int Ndim, int activation) {
  const int lane  = threadIdx.x & 31;
  // wave-uniform tile coordinate, proven uniform to the compiler:
  const int nbase = __builtin_amdgcn_readfirstlane(
      (blockIdx.y * 4 + (threadIdx.x >> 5)) * 64);
  const int m0 = blockIdx.x * 16;
  if (nbase >= Ndim) return;                 // scalar branch

  const v8f vzero = {0.f, 0.f, 0.f, 0.f, 0.f, 0.f, 0.f, 0.f};
  v8f acc[4];
#pragma unroll
  for (int j = 0; j < 4; ++j) acc[j] = vzero;

  if (nbase + 64 <= Ndim) {                  // fast path: full tile, pipelined
    gemm_panel_fast(A1, B1, K1, m0, nbase, lane, acc);
    if (A2 != nullptr) gemm_panel_fast(A2, B2, K2, m0, nbase, lane, acc);
  } else {                                   // edge tile (only last tile of N=600)
    gemm_panel_edge(A1, B1, K1, m0, nbase, lane, Ndim, acc);
    if (A2 != nullptr) gemm_panel_edge(A2, B2, K2, m0, nbase, lane, Ndim, acc);
  }

  // Epilogue: C/D layout — lane L: col = n0+(L&15), rows = m0 + (L>>4)*8 + i
  const int rbase = m0 + ((lane >> 4) << 3);
#pragma unroll
  for (int j = 0; j < 4; ++j) {
    const int col = nbase + j * 16 + (lane & 15);
    if (col < Ndim) {                        // store-only divergence: fine
      const float bval = bias ? bias[col] : 0.f;
#pragma unroll
      for (int i = 0; i < 8; ++i) {
        const int row = rbase + i;
        float v = acc[j][i] + bval + (rowoff ? rowoff[row] : 0.f);
        if (activation == 1) v = tanhf(v);
        C[(size_t)row * Ndim + col] = v;
        if (Cb != nullptr)
          ((unsigned short*)Cb)[(size_t)row * Ndim + col] = f32_to_bf16_bits(v);
      }
    }
  }
}

// ---- Elementwise / prep kernels ---------------------------------------------
__global__ void __launch_bounds__(256)
k_cvt_bf16(const float* __restrict__ src, bf16_t* __restrict__ dst, int n) {
  int i = blockIdx.x * 256 + threadIdx.x;
  if (i < n) ((unsigned short*)dst)[i] = f32_to_bf16_bits(src[i]);
}

__global__ void __launch_bounds__(256)
k_transpose_cvt(const float* __restrict__ U, bf16_t* __restrict__ UT, int Kdim, int Ndim) {
  int idx = blockIdx.x * 256 + threadIdx.x;      // U is Kdim x Ndim; UT is Ndim x Kdim
  if (idx >= Kdim * Ndim) return;
  int k = idx / Ndim, n = idx % Ndim;
  ((unsigned short*)UT)[(size_t)n * Kdim + k] = f32_to_bf16_bits(U[idx]);
}

__global__ void __launch_bounds__(256)
k_bias_sum(const float* __restrict__ a, const float* __restrict__ b,
           float* __restrict__ o, int n) {
  int i = blockIdx.x * 256 + threadIdx.x;
  if (i < n) o[i] = a[i] + b[i];
}

// Gather parent/sibling states; also emit bf16 mirrors of h_par / h_sib
// (the WMMA A-operands) so the GEMM loop does zero conversion work.
__global__ void __launch_bounds__(256)
k_gather(const float* __restrict__ z, const int* __restrict__ pa_t,
         const int* __restrict__ si_t,
         const float* __restrict__ h_p, const float* __restrict__ c_p,
         const float* __restrict__ h_s, const float* __restrict__ c_s,
         float* __restrict__ h_par, float* __restrict__ c_par,
         float* __restrict__ h_sib, float* __restrict__ c_sib,
         bf16_t* __restrict__ h_par_b, bf16_t* __restrict__ h_sib_b) {
  const int r = blockIdx.x, d = threadIdx.x;
  const int pa = pa_t[r], si = si_t[r];
  const size_t o = (size_t)r * DDIM + d;
  const float hp = (pa < 0) ? z[o] : h_p[(size_t)pa * DDIM + d];
  const float hs = (si < 0) ? 0.f : h_s[(size_t)si * DDIM + d];
  h_par[o] = hp;
  c_par[o] = (pa < 0) ? 0.f : c_p[(size_t)pa * DDIM + d];
  h_sib[o] = hs;
  c_sib[o] = (si < 0) ? 0.f : c_s[(size_t)si * DDIM + d];
  ((unsigned short*)h_par_b)[o] = f32_to_bf16_bits(hp);
  ((unsigned short*)h_sib_b)[o] = f32_to_bf16_bits(hs);
}

__device__ __forceinline__ float softplusf(float x) {
  return fmaxf(x, 0.f) + log1pf(__expf(-fabsf(x)));
}
__device__ __forceinline__ float sigf(float x) { return 1.f / (1.f + __expf(-x)); }

__global__ void __launch_bounds__(64)
k_head(const float* __restrict__ h_pred,
       const float* __restrict__ Wd, const float* __restrict__ bd,
       const float* __restrict__ Ww, const float* __restrict__ bw,
       const float* __restrict__ isp_t, const float* __restrict__ hsb_t,
       const float* __restrict__ w_offp, const float* __restrict__ w_offs,
       float* __restrict__ off, float* __restrict__ loss) {
  const int r = blockIdx.x * 64 + threadIdx.x;
  const float* h = h_pred + (size_t)r * DDIM;
  float dl = bd[0], wl = bw[0];
  for (int k = 0; k < DDIM; ++k) { dl += h[k] * Wd[k]; wl += h[k] * Ww[k]; }
  const float isp = isp_t[r], hsb = hsb_t[r];
  atomicAdd(loss, (softplusf(dl) - dl * isp) + (softplusf(wl) - wl * hsb));
  off[r] = isp * w_offp[0] + hsb * w_offs[0];
}

__global__ void __launch_bounds__(256)
k_vocab_loss(const float* __restrict__ logits_res, const float* __restrict__ logits_name,
             const int* __restrict__ lab_t, const int* __restrict__ tr_t,
             float* __restrict__ loss) {
  __shared__ float sred[256];
  const int r = blockIdx.x, tid = threadIdx.x;
  const bool isres = (tr_t[r] == 1);
  const float* lg = isres ? (logits_res + (size_t)r * VRDIM)
                          : (logits_name + (size_t)r * VNDIM);
  const int V = isres ? VRDIM : VNDIM;
  int lab = lab_t[r]; lab = lab < 0 ? 0 : (lab >= V ? V - 1 : lab);
  float mx = -3.4e38f;
  for (int i = tid; i < V; i += 256) mx = fmaxf(mx, lg[i]);
  sred[tid] = mx; __syncthreads();
  for (int s = 128; s > 0; s >>= 1) {
    if (tid < s) sred[tid] = fmaxf(sred[tid], sred[tid + s]);
    __syncthreads();
  }
  mx = sred[0]; __syncthreads();
  float se = 0.f;
  for (int i = tid; i < V; i += 256) se += __expf(lg[i] - mx);
  sred[tid] = se; __syncthreads();
  for (int s = 128; s > 0; s >>= 1) {
    if (tid < s) sred[tid] += sred[tid + s];
    __syncthreads();
  }
  if (tid == 0) atomicAdd(loss, (mx + __logf(sred[0])) - lg[lab]);
}

// Teacher-forced embedding gather, written directly as bf16 (GEMM A-operand).
__global__ void __launch_bounds__(256)
k_embed(const int* __restrict__ lab_t, const float* __restrict__ emb_res,
        const float* __restrict__ emb_name, bf16_t* __restrict__ x_r_b,
        bf16_t* __restrict__ x_n_b) {
  const int r = blockIdx.x, d = threadIdx.x;
  int lab = lab_t[r]; if (lab < 0) lab = 0;
  const int lr = lab >= VRDIM ? VRDIM - 1 : lab;
  const int ln = lab >= VNDIM ? VNDIM - 1 : lab;
  ((unsigned short*)x_r_b)[(size_t)r * DDIM + d] =
      f32_to_bf16_bits(emb_res[(size_t)lr * DDIM + d]);
  if (d < ELDIM)
    ((unsigned short*)x_n_b)[(size_t)r * ELDIM + d] =
        f32_to_bf16_bits(emb_name[(size_t)ln * ELDIM + d]);
}

__global__ void __launch_bounds__(256)
k_lstm_update(const float* __restrict__ gates_p, const float* __restrict__ gates_s,
              const float* __restrict__ gates_l,
              const float* __restrict__ c_par, const float* __restrict__ c_sib,
              const int* __restrict__ tr_t, int t,
              float* __restrict__ h_p, float* __restrict__ c_p,
              float* __restrict__ h_s, float* __restrict__ c_s) {
  const int r = blockIdx.x, d = threadIdx.x;
  const size_t gb = (size_t)r * (4 * DDIM);
  float hp2, cp2, hs2, cs2, hl2, cl2;
  {
    const float* g = gates_p;
    float i = sigf(g[gb + d]), f = sigf(g[gb + DDIM + d]);
    float gg = tanhf(g[gb + 2 * DDIM + d]), o = sigf(g[gb + 3 * DDIM + d]);
    cp2 = f * c_par[(size_t)r * DDIM + d] + i * gg; hp2 = o * tanhf(cp2);
  }
  {
    const float* g = gates_s;
    float i = sigf(g[gb + d]), f = sigf(g[gb + DDIM + d]);
    float gg = tanhf(g[gb + 2 * DDIM + d]), o = sigf(g[gb + 3 * DDIM + d]);
    cs2 = f * c_sib[(size_t)r * DDIM + d] + i * gg; hs2 = o * tanhf(cs2);
  }
  {
    const float* g = gates_l;
    float i = sigf(g[gb + d]), f = sigf(g[gb + DDIM + d]);
    float gg = tanhf(g[gb + 2 * DDIM + d]), o = sigf(g[gb + 3 * DDIM + d]);
    cl2 = f * c_sib[(size_t)r * DDIM + d] + i * gg; hl2 = o * tanhf(cl2);
  }
  const bool res = (tr_t[r] == 1);
  const size_t id = ((size_t)t * MROWS + r) * DDIM + d;
  h_p[id] = res ? hp2 : 0.f;  c_p[id] = res ? cp2 : 0.f;
  h_s[id] = res ? hs2 : hl2;  c_s[id] = res ? cs2 : cl2;
}

__global__ void k_finalize(const float* __restrict__ loss, float* __restrict__ out) {
  if (threadIdx.x == 0 && blockIdx.x == 0) out[0] = loss[0] / (float)NNODES;
}

// ---- Host orchestration ------------------------------------------------------
extern "C" void kernel_launch(void* const* d_in, const int* in_sizes, int n_in,
                              void* d_out, int out_size, void* d_ws, size_t ws_size,
                              hipStream_t stream) {
  (void)in_sizes; (void)n_in; (void)out_size; (void)ws_size;
  const float* z        = (const float*)d_in[0];
  const float* U_p      = (const float*)d_in[1];
  const float* U_s      = (const float*)d_in[2];
  const float* Wd       = (const float*)d_in[3];
  const float* bd       = (const float*)d_in[4];
  const float* Ww       = (const float*)d_in[5];
  const float* bw       = (const float*)d_in[6];
  const float* Wih_p    = (const float*)d_in[7];
  const float* Whh_p    = (const float*)d_in[8];
  const float* bih_p    = (const float*)d_in[9];
  const float* bhh_p    = (const float*)d_in[10];
  const float* Wih_s    = (const float*)d_in[11];
  const float* Whh_s    = (const float*)d_in[12];
  const float* bih_s    = (const float*)d_in[13];
  const float* bhh_s    = (const float*)d_in[14];
  const float* Wih_l    = (const float*)d_in[15];
  const float* Whh_l    = (const float*)d_in[16];
  const float* bih_l    = (const float*)d_in[17];
  const float* bhh_l    = (const float*)d_in[18];
  const float* predWr   = (const float*)d_in[19];
  const float* predbr   = (const float*)d_in[20];
  const float* predWn   = (const float*)d_in[21];
  const float* predbn   = (const float*)d_in[22];
  const float* emb_res  = (const float*)d_in[23];
  const float* emb_name = (const float*)d_in[24];
  const float* w_offp   = (const float*)d_in[25];
  const float* w_offs   = (const float*)d_in[26];
  const float* isp      = (const float*)d_in[27];
  const float* hsb      = (const float*)d_in[28];
  const int* parent_idx = (const int*)d_in[29];
  const int* sib_idx    = (const int*)d_in[30];
  const int* labels     = (const int*)d_in[31];
  const int* type_res   = (const int*)d_in[32];

  char* p = (char*)d_ws;
  auto carve = [&](size_t bytes) -> void* {
    void* r = (void*)p;
    p += (bytes + 255) & ~(size_t)255;
    return r;
  };

  const size_t stateElems = (size_t)NNODES * DDIM;
  float* h_p = (float*)carve(4 * sizeof(float) * stateElems);   // h_p,c_p,h_s,c_s contiguous
  float* c_p = h_p + stateElems;
  float* h_s = c_p + stateElems;
  float* c_s = h_s + stateElems;

  float* h_par  = (float*)carve(sizeof(float) * MROWS * DDIM);
  float* c_par  = (float*)carve(sizeof(float) * MROWS * DDIM);
  float* h_sib  = (float*)carve(sizeof(float) * MROWS * DDIM);
  float* c_sib  = (float*)carve(sizeof(float) * MROWS * DDIM);
  float* h_pred = (float*)carve(sizeof(float) * MROWS * DDIM);
  float* gates_p = (float*)carve(sizeof(float) * MROWS * 4 * DDIM);
  float* gates_s = (float*)carve(sizeof(float) * MROWS * 4 * DDIM);
  float* gates_l = (float*)carve(sizeof(float) * MROWS * 4 * DDIM);
  float* logits_res  = (float*)carve(sizeof(float) * MROWS * VRDIM);
  float* logits_name = (float*)carve(sizeof(float) * MROWS * VNDIM);
  float* off  = (float*)carve(sizeof(float) * MROWS);
  float* loss = (float*)carve(256);

  // bf16 activation mirrors (WMMA A-operands)
  bf16_t* h_par_b  = (bf16_t*)carve(2 * MROWS * DDIM);
  bf16_t* h_sib_b  = (bf16_t*)carve(2 * MROWS * DDIM);
  bf16_t* h_pred_b = (bf16_t*)carve(2 * MROWS * DDIM);
  bf16_t* x_r_b    = (bf16_t*)carve(2 * MROWS * DDIM);
  bf16_t* x_n_b    = (bf16_t*)carve(2 * MROWS * ELDIM);

  // bf16 weight panels (WMMA B-operands, N x K)
  bf16_t* UpT     = (bf16_t*)carve(2 * DDIM * DDIM);
  bf16_t* UsT     = (bf16_t*)carve(2 * DDIM * DDIM);
  bf16_t* Wih_p_b = (bf16_t*)carve(2 * 4 * DDIM * DDIM);
  bf16_t* Whh_p_b = (bf16_t*)carve(2 * 4 * DDIM * DDIM);
  bf16_t* Wih_s_b = (bf16_t*)carve(2 * 4 * DDIM * DDIM);
  bf16_t* Whh_s_b = (bf16_t*)carve(2 * 4 * DDIM * DDIM);
  bf16_t* Whh_l_b = (bf16_t*)carve(2 * 4 * DDIM * DDIM);
  bf16_t* Wih_l_b = (bf16_t*)carve(2 * 4 * DDIM * ELDIM);
  bf16_t* pWr_b   = (bf16_t*)carve(2 * (size_t)VRDIM * DDIM);
  bf16_t* pWn_b   = (bf16_t*)carve(2 * (size_t)VNDIM * DDIM);
  float* bsum_p = (float*)carve(sizeof(float) * 4 * DDIM);
  float* bsum_s = (float*)carve(sizeof(float) * 4 * DDIM);
  float* bsum_l = (float*)carve(sizeof(float) * 4 * DDIM);

  // Zero the scan state and the loss accumulator (graph-capture legal).
  hipMemsetAsync(h_p, 0, 4 * sizeof(float) * stateElems, stream);
  hipMemsetAsync(loss, 0, sizeof(float), stream);

  // One-time weight prep: transpose U to N x K; convert all B operands to bf16.
  const int uElems = DDIM * DDIM;
  k_transpose_cvt<<<(uElems + 255) / 256, 256, 0, stream>>>(U_p, UpT, DDIM, DDIM);
  k_transpose_cvt<<<(uElems + 255) / 256, 256, 0, stream>>>(U_s, UsT, DDIM, DDIM);
  const int wElems = 4 * DDIM * DDIM;
  k_cvt_bf16<<<(wElems + 255) / 256, 256, 0, stream>>>(Wih_p, Wih_p_b, wElems);
  k_cvt_bf16<<<(wElems + 255) / 256, 256, 0, stream>>>(Whh_p, Whh_p_b, wElems);
  k_cvt_bf16<<<(wElems + 255) / 256, 256, 0, stream>>>(Wih_s, Wih_s_b, wElems);
  k_cvt_bf16<<<(wElems + 255) / 256, 256, 0, stream>>>(Whh_s, Whh_s_b, wElems);
  k_cvt_bf16<<<(wElems + 255) / 256, 256, 0, stream>>>(Whh_l, Whh_l_b, wElems);
  const int wlElems = 4 * DDIM * ELDIM;
  k_cvt_bf16<<<(wlElems + 255) / 256, 256, 0, stream>>>(Wih_l, Wih_l_b, wlElems);
  const int prElems = VRDIM * DDIM;
  k_cvt_bf16<<<(prElems + 255) / 256, 256, 0, stream>>>(predWr, pWr_b, prElems);
  const int pnElems = VNDIM * DDIM;
  k_cvt_bf16<<<(pnElems + 255) / 256, 256, 0, stream>>>(predWn, pWn_b, pnElems);
  k_bias_sum<<<4, 256, 0, stream>>>(bih_p, bhh_p, bsum_p, 4 * DDIM);
  k_bias_sum<<<4, 256, 0, stream>>>(bih_s, bhh_s, bsum_s, 4 * DDIM);
  k_bias_sum<<<4, 256, 0, stream>>>(bih_l, bhh_l, bsum_l, 4 * DDIM);

  const int mt = MROWS / 16;  // 32 M-tiles
  for (int t = 0; t < TSTEPS; ++t) {
    const int* pa_t   = parent_idx + t * MROWS;
    const int* si_t   = sib_idx + t * MROWS;
    const int* lab_t  = labels + t * MROWS;
    const int* tr_t   = type_res + t * MROWS;
    const float* isp_t = isp + t * MROWS;
    const float* hsb_t = hsb + t * MROWS;

    k_gather<<<MROWS, DDIM, 0, stream>>>(z, pa_t, si_t, h_p, c_p, h_s, c_s,
                                         h_par, c_par, h_sib, c_sib,
                                         h_par_b, h_sib_b);

    // h_pred = tanh(h_par@U_p + h_sib@U_s); also writes bf16 mirror
    k_gemm<<<dim3(mt, 1), 128, 0, stream>>>(h_par_b, UpT, DDIM, h_sib_b, UsT, DDIM,
                                            nullptr, nullptr, h_pred, h_pred_b,
                                            MROWS, DDIM, 1);

    k_head<<<MROWS / 64, 64, 0, stream>>>(h_pred, Wd, bd, Ww, bw, isp_t, hsb_t,
                                          w_offp, w_offs, off, loss);

    // vocab logits (bias + per-row offset fused)
    k_gemm<<<dim3(mt, (VRDIM + 255) / 256), 128, 0, stream>>>(
        h_pred_b, pWr_b, DDIM, nullptr, nullptr, 0, predbr, off,
        logits_res, nullptr, MROWS, VRDIM, 0);
    k_gemm<<<dim3(mt, (VNDIM + 255) / 256), 128, 0, stream>>>(
        h_pred_b, pWn_b, DDIM, nullptr, nullptr, 0, predbn, off,
        logits_name, nullptr, MROWS, VNDIM, 0);

    k_vocab_loss<<<MROWS, 256, 0, stream>>>(logits_res, logits_name, lab_t, tr_t, loss);

    k_embed<<<MROWS, 256, 0, stream>>>(lab_t, emb_res, emb_name, x_r_b, x_n_b);

    // LSTM gates: x@Wih^T + h@Whh^T + (bih+bhh)
    k_gemm<<<dim3(mt, 4), 128, 0, stream>>>(x_r_b, Wih_p_b, DDIM, h_par_b, Whh_p_b, DDIM,
                                            bsum_p, nullptr, gates_p, nullptr,
                                            MROWS, 4 * DDIM, 0);
    k_gemm<<<dim3(mt, 4), 128, 0, stream>>>(x_r_b, Wih_s_b, DDIM, h_sib_b, Whh_s_b, DDIM,
                                            bsum_s, nullptr, gates_s, nullptr,
                                            MROWS, 4 * DDIM, 0);
    k_gemm<<<dim3(mt, 4), 128, 0, stream>>>(x_n_b, Wih_l_b, ELDIM, h_sib_b, Whh_l_b, DDIM,
                                            bsum_l, nullptr, gates_l, nullptr,
                                            MROWS, 4 * DDIM, 0);

    k_lstm_update<<<MROWS, 256, 0, stream>>>(gates_p, gates_s, gates_l, c_par, c_sib,
                                             tr_t, t, h_p, c_p, h_s, c_s);
  }

  k_finalize<<<1, 32, 0, stream>>>(loss, (float*)d_out);
}
